// MACE_38104949850359
// MI455X (gfx1250) — matrix operand order repaired and checked by reading, not compiled, not run on hardware
//
#include <hip/hip_runtime.h>
#include <math.h>

// ---------------------------------------------------------------------------
// MACE-like GNN on gfx1250 (CDNA5). bf16 WMMA (v_wmma_f32_16x16x32_bf16) for
// the edge MLP and the channel-mix GEMMs; f32 atomics into an L2-resident
// accumulator for the edge->node segment sum; VALU for small per-node algebra.
//
// WMMA fragment layouts (per CDNA5 ISA 7.12.2):
//   A (16x32 bf16): lane m=lane&15, hs=lane>>4; halves0-7 = K[hs*8 .. +7],
//                   halves8-15 = K[16+hs*8 .. +7], row m.
//   B (32x16 bf16): lane n=lane&15, hs=lane>>4; halves0-7 = K[hs*8 .. +7], halves8-15 = K[16+hs*8 .. +7],
//                   col n.  (B matrices stored column-major in LDS so each
//                   lane's 16 elements sit in two contiguous runs of 8.)
//   C/D (16x16 f32): lane n=lane&15; VGPR r -> element (row r+8*hs, col n).
// ---------------------------------------------------------------------------

typedef __bf16 bf16;
typedef __attribute__((ext_vector_type(16))) __bf16 v16bf;
typedef __attribute__((ext_vector_type(8)))  float  v8f;

__device__ __forceinline__ v8f vzero8() {
  v8f v;
#pragma unroll
  for (int i = 0; i < 8; ++i) v[i] = 0.f;
  return v;
}

__device__ __forceinline__ v8f wmma_bf16(v16bf a, v16bf b, v8f c) {
  return __builtin_amdgcn_wmma_f32_16x16x32_bf16(
      /*neg_a=*/false, a, /*neg_b=*/false, b,
      /*c_mod=*/(short)0, c, /*reuse_a=*/false, /*reuse_b=*/false);
}

// A fragment from row-major bf16 tile (16 rows, stride lda), K-slice at k0.
__device__ __forceinline__ v16bf load_a_row(const bf16* src, int lda, int k0, int lane) {
  const int m = lane & 15, hs = lane >> 4;
  const bf16* p = src + m * lda + k0 + hs * 8;
  v16bf a;
#pragma unroll
  for (int i = 0; i < 8; ++i) { a[i] = p[i]; a[8 + i] = p[16 + i]; }
  return a;
}

// B fragment from column-major bf16 tile (srcT[col*ldk + k]), K-slice at k0,
// 16 columns starting at col0.
__device__ __forceinline__ v16bf load_b_cm(const bf16* srcT, int ldk, int k0, int col0, int lane) {
  const int n = lane & 15, hs = lane >> 4;
  const bf16* p = srcT + (col0 + n) * ldk + k0 + hs * 8;
  v16bf b;
#pragma unroll
  for (int i = 0; i < 8; ++i) { b[i] = p[i]; b[8 + i] = p[16 + i]; }
  return b;
}

// Fast silu: result feeds a bf16 quantization, so a 1-ulp v_rcp_f32 is plenty.
__device__ __forceinline__ float silu(float x) {
  return x * __builtin_amdgcn_rcpf(1.f + __expf(-x));
}

// ---------------------------------------------------------------------------
__global__ void mace_zero4_kernel(float4* __restrict__ p, size_t n4) {
  size_t i = (size_t)blockIdx.x * blockDim.x + threadIdx.x;
  const size_t st = (size_t)gridDim.x * blockDim.x;
  const float4 z = make_float4(0.f, 0.f, 0.f, 0.f);
  for (; i < n4; i += st) p[i] = z;
}

// z[n], feats[n]=W_embed[z], energy[batch[n]] += ae_w[z]
__global__ __launch_bounds__(256) void mace_setup_kernel(
    const float* __restrict__ attrs, const float* __restrict__ Wemb,
    const float* __restrict__ aew, const int* __restrict__ batch,
    float* __restrict__ feats, int* __restrict__ z,
    float* __restrict__ energy, int N) {
  const int t = blockIdx.x * 256 + threadIdx.x;
  if (t >= N * 64) return;
  const int n = t >> 6, j = t & 63;
  const float* a = attrs + (size_t)n * 10;
  int zz = 0;
#pragma unroll
  for (int c = 0; c < 10; ++c)
    if (a[c] > 0.5f) zz = c;
  feats[t] = Wemb[zz * 64 + j];
  if (j == 0) {
    z[n] = zz;
    atomicAdd(&energy[batch[n]], aew[zz]);
  }
}

// Per-edge geometry: Y (16, f32) and radial basis (8, bf16).
__global__ __launch_bounds__(256) void mace_geom_kernel(
    const float* __restrict__ vec, float* __restrict__ Yg,
    bf16* __restrict__ efb, int E) {
  const int e = blockIdx.x * 256 + threadIdx.x;
  if (e >= E) return;
  const float vx = vec[3 * (size_t)e + 0];
  const float vy = vec[3 * (size_t)e + 1];
  const float vz = vec[3 * (size_t)e + 2];
  const float r = sqrtf(vx * vx + vy * vy + vz * vz);
  const float ir = __builtin_amdgcn_rcpf(r);
  const float x = vx * ir, y = vy * ir, zc = vz * ir;
  const float x2 = x * x, y2 = y * y, z2 = zc * zc;
  const float s3 = 1.7320508075688772f, s5 = 2.23606797749979f, s15 = 3.872983346207417f;
  const float c9 = 2.091650066335189f;    // sqrt(35/8)
  const float c10 = 10.246950765959598f;  // sqrt(105)
  const float c11 = 1.6201851746019651f;  // sqrt(21/8)
  const float c12 = 1.3228756555322954f;  // 0.5*sqrt(7)
  const float c14 = 5.123475382979799f;   // 0.5*sqrt(105)
  float Yl[16];
  Yl[0] = 1.f;
  Yl[1] = s3 * x; Yl[2] = s3 * y; Yl[3] = s3 * zc;
  Yl[4] = s15 * x * y; Yl[5] = s15 * y * zc;
  Yl[6] = 0.5f * s5 * (3.f * z2 - 1.f);
  Yl[7] = s15 * x * zc; Yl[8] = 0.5f * s15 * (x2 - y2);
  Yl[9] = c9 * y * (3.f * x2 - y2);
  Yl[10] = c10 * x * y * zc;
  Yl[11] = c11 * y * (5.f * z2 - 1.f);
  Yl[12] = c12 * zc * (5.f * z2 - 3.f);
  Yl[13] = c11 * x * (5.f * z2 - 1.f);
  Yl[14] = c14 * zc * (x2 - y2);
  Yl[15] = c9 * x * (x2 - 3.f * y2);
  float4* o = (float4*)(Yg + (size_t)e * 16);
  o[0] = make_float4(Yl[0], Yl[1], Yl[2], Yl[3]);
  o[1] = make_float4(Yl[4], Yl[5], Yl[6], Yl[7]);
  o[2] = make_float4(Yl[8], Yl[9], Yl[10], Yl[11]);
  o[3] = make_float4(Yl[12], Yl[13], Yl[14], Yl[15]);
  const float xr = r * 0.2f;  // r / R_MAX
  float env = 0.f;
  if (xr < 1.f) {
    const float xr2 = xr * xr;
    const float xr5 = xr2 * xr2 * xr;
    env = 1.f + xr5 * (-21.f + xr * (35.f - 15.f * xr));
  }
  const float s = 0.6324555320336759f /* sqrt(2/5) */ * ir * env;
#pragma unroll
  for (int nb = 1; nb <= 8; ++nb)
    efb[(size_t)e * 8 + nb - 1] =
        (bf16)(s * __sinf(3.14159265358979323846f * (float)nb * xr));
}

// ---------------------------------------------------------------------------
// Fused edge MLP (WMMA) + message formation + scatter-add into Apre[s][n][k].
// One 16-edge tile per wave; 4 waves / 128-thread block; all weights staged
// (bf16, column-major) in LDS.
// ---------------------------------------------------------------------------
__global__ __launch_bounds__(128) void mace_edge_kernel(
    const float* __restrict__ R0g, const float* __restrict__ R1g,
    const float* __restrict__ R2g, const float* __restrict__ R3g,
    const bf16* __restrict__ efb, const float* __restrict__ Yg,
    const float* __restrict__ feats, const int* __restrict__ snd,
    const int* __restrict__ rcv, float* __restrict__ Apre,
    int E, int N) {
  __shared__ bf16 sR0t[64 * 8];     // [col][k], k<8
  __shared__ bf16 sR1t[64 * 64];    // [col][k]
  __shared__ bf16 sR2t[64 * 64];
  __shared__ bf16 sR3t[256 * 64];   // [col][k]
  __shared__ bf16 sH[4][16 * 64];   // per-wave activations, row-major [edge][feat]
  __shared__ float sY[4][16 * 16];
  __shared__ int sSnd[4][16];
  __shared__ int sRcv[4][16];

  const int tid = threadIdx.x;
  for (int idx = tid; idx < 64 * 8; idx += 128) {
    const int c = idx >> 3, k = idx & 7;
    sR0t[idx] = (bf16)R0g[k * 64 + c];
  }
  for (int idx = tid; idx < 64 * 64; idx += 128) {
    const int c = idx >> 6, k = idx & 63;
    sR1t[idx] = (bf16)R1g[k * 64 + c];
    sR2t[idx] = (bf16)R2g[k * 64 + c];
  }
  for (int idx = tid; idx < 256 * 64; idx += 128) {
    const int c = idx >> 6, k = idx & 63;
    sR3t[idx] = (bf16)R3g[k * 256 + c];
  }
  __syncthreads();

  const int w = tid >> 5, lane = tid & 31;
  const int tiles = E >> 4;
  const int wt = blockIdx.x * 4 + w;
  if (wt >= tiles) return;
  const int e0 = wt << 4;
  const int mrow = lane & 15, hs = lane >> 4;

  if (lane < 16) {
    sSnd[w][lane] = snd[e0 + lane];
    sRcv[w][lane] = rcv[e0 + lane];
  }
  for (int t = lane; t < 256; t += 32) sY[w][t] = Yg[(size_t)e0 * 16 + t];
  __builtin_amdgcn_wave_barrier();

  // ef A-fragment: true K=8, zero-padded to 32.
  v16bf fa;
#pragma unroll
  for (int i = 0; i < 16; ++i) fa[i] = (bf16)0.f;
  if (hs == 0) {
    const bf16* p = efb + (size_t)(e0 + mrow) * 8;
#pragma unroll
    for (int i = 0; i < 8; ++i) fa[i] = p[i];
  }

  // Stage 1: h = silu(ef @ R0)  (16x64)
#pragma unroll
  for (int ct = 0; ct < 4; ++ct) {
    v16bf b;
#pragma unroll
    for (int i = 0; i < 16; ++i) b[i] = (bf16)0.f;
    if (hs == 0) {  // only K<8 rows are non-zero; they live in lanes 0-15
      const bf16* p = sR0t + (ct * 16 + mrow) * 8;
#pragma unroll
      for (int i = 0; i < 8; ++i) b[i] = p[i];
    }
    v8f acc = wmma_bf16(fa, b, vzero8());
#pragma unroll
    for (int r = 0; r < 8; ++r)
      sH[w][(r + 8 * hs) * 64 + ct * 16 + mrow] = (bf16)silu(acc[r]);
  }
  __builtin_amdgcn_wave_barrier();

  // Stage 2: h = silu(h @ R1)
  {
    v16bf a0 = load_a_row(sH[w], 64, 0, lane);
    v16bf a1 = load_a_row(sH[w], 64, 32, lane);
    __builtin_amdgcn_wave_barrier();
#pragma unroll
    for (int ct = 0; ct < 4; ++ct) {
      v8f acc = wmma_bf16(a0, load_b_cm(sR1t, 64, 0, ct * 16, lane), vzero8());
      acc = wmma_bf16(a1, load_b_cm(sR1t, 64, 32, ct * 16, lane), acc);
#pragma unroll
      for (int r = 0; r < 8; ++r)
        sH[w][(r + 8 * hs) * 64 + ct * 16 + mrow] = (bf16)silu(acc[r]);
    }
  }
  __builtin_amdgcn_wave_barrier();

  // Stage 3: h = silu(h @ R2)
  {
    v16bf a0 = load_a_row(sH[w], 64, 0, lane);
    v16bf a1 = load_a_row(sH[w], 64, 32, lane);
    __builtin_amdgcn_wave_barrier();
#pragma unroll
    for (int ct = 0; ct < 4; ++ct) {
      v8f acc = wmma_bf16(a0, load_b_cm(sR2t, 64, 0, ct * 16, lane), vzero8());
      acc = wmma_bf16(a1, load_b_cm(sR2t, 64, 32, ct * 16, lane), acc);
#pragma unroll
      for (int r = 0; r < 8; ++r)
        sH[w][(r + 8 * hs) * 64 + ct * 16 + mrow] = (bf16)silu(acc[r]);
    }
  }
  __builtin_amdgcn_wave_barrier();

  // Stage 4: W = h @ R3 (16x256), consumed tile-by-tile into the scatter.
  {
    v16bf a0 = load_a_row(sH[w], 64, 0, lane);
    v16bf a1 = load_a_row(sH[w], 64, 32, lane);
    const size_t NS = (size_t)N * 64;
#pragma unroll
    for (int ct = 0; ct < 16; ++ct) {
      v8f acc = wmma_bf16(a0, load_b_cm(sR3t, 64, 0, ct * 16, lane), vzero8());
      acc = wmma_bf16(a1, load_b_cm(sR3t, 64, 32, ct * 16, lane), acc);
      const int col = ct * 16 + mrow;   // column index into K*4
      const int kf = col >> 2;          // channel
      const int l = col & 3;            // angular degree
      const int sb = l * l;             // first spherical index of degree l
      const int scnt = 2 * l + 1;
#pragma unroll
      for (int r = 0; r < 8; ++r) {
        const int e = r + 8 * hs;
        const float xs = feats[(size_t)sSnd[w][e] * 64 + kf];
        const float wx = acc[r] * xs;
        float* dst = Apre + (size_t)sRcv[w][e] * 64 + kf;
        for (int s = sb; s < sb + scnt; ++s)
          atomicAdd(dst + (size_t)s * NS, wx * sY[w][e * 16 + s]);
      }
    }
  }
}

// ---------------------------------------------------------------------------
// Channel mix: Amix[m][n][j] = (1/16) * sum_k Apre[m][n][k] * Wmix[lidx[m]][k][j]
// ---------------------------------------------------------------------------
__global__ __launch_bounds__(256) void mace_mix_kernel(
    const float* __restrict__ Apre, const float* __restrict__ WmixL,
    float* __restrict__ Amix, int N) {
  __shared__ bf16 sBt[64 * 64];  // [j][k]
  const int mm = blockIdx.y;
  const int l = (mm == 0) ? 0 : (mm < 4) ? 1 : (mm < 9) ? 2 : 3;
  const int tid = threadIdx.x;
  for (int idx = tid; idx < 4096; idx += 256) {
    const int j = idx >> 6, k = idx & 63;
    sBt[idx] = (bf16)WmixL[l * 4096 + k * 64 + j];
  }
  __syncthreads();

  const int w = tid >> 5, lane = tid & 31;
  const int tiles = (N + 15) >> 4;
  const int wt = blockIdx.x * 8 + w;
  if (wt >= tiles) return;
  const int row0 = wt << 4;
  const int mr = lane & 15, hs = lane >> 4;
  const size_t MS = (size_t)N * 64;
  const float* prow = Apre + (size_t)mm * MS + (size_t)(row0 + mr) * 64;
  v16bf a0, a1;
#pragma unroll
  for (int i = 0; i < 8; ++i) {
    a0[i] = (bf16)prow[hs * 8 + i];
    a0[8 + i] = (bf16)prow[16 + hs * 8 + i];
    a1[i] = (bf16)prow[32 + hs * 8 + i];
    a1[8 + i] = (bf16)prow[48 + hs * 8 + i];
  }
  float* Om = Amix + (size_t)mm * MS;
#pragma unroll
  for (int ct = 0; ct < 4; ++ct) {
    v8f acc = wmma_bf16(a0, load_b_cm(sBt, 64, 0, ct * 16, lane), vzero8());
    acc = wmma_bf16(a1, load_b_cm(sBt, 64, 32, ct * 16, lane), acc);
#pragma unroll
    for (int r = 0; r < 8; ++r)
      Om[(size_t)(row0 + r + 8 * hs) * 64 + ct * 16 + mr] = acc[r] * 0.0625f;  // 1/AVG_NEIGH
  }
}

// ---------------------------------------------------------------------------
// Per-node invariants + feature update + readout. One wave per node.
// ---------------------------------------------------------------------------
__global__ __launch_bounds__(256) void mace_node_kernel(
    const float* __restrict__ Amix, float* __restrict__ feats,
    const int* __restrict__ z, const int* __restrict__ batch,
    const float* __restrict__ U2, const float* __restrict__ U3,
    const float* __restrict__ Wsc, const float* __restrict__ Winv1,
    const float* __restrict__ Wread0, const float* __restrict__ Wr1,
    const float* __restrict__ Wr2, float* __restrict__ energy,
    int N, int is_last) {
  __shared__ float sF[8][64];
  const int w = threadIdx.x >> 5, lane = threadIdx.x & 31;
  const int n = blockIdx.x * 8 + w;
  if (n >= N) return;
  const int zz = z[n];
  const int j0 = lane, j1 = lane + 32;
  const size_t base = (size_t)n * 64;
  const size_t MS = (size_t)N * 64;

  float p0[4] = {0.f, 0.f, 0.f, 0.f}, p1[4] = {0.f, 0.f, 0.f, 0.f};
  float i10 = 0.f, i11 = 0.f;
#pragma unroll
  for (int mm = 0; mm < 16; ++mm) {
    const int l = (mm == 0) ? 0 : (mm < 4) ? 1 : (mm < 9) ? 2 : 3;
    const float v0 = Amix[(size_t)mm * MS + base + j0];
    const float v1 = Amix[(size_t)mm * MS + base + j1];
    if (mm == 0) { i10 = v0; i11 = v1; }
    p0[l] += v0 * v0;
    p1[l] += v1 * v1;
  }
  const float* u2a = U2 + ((size_t)zz * 64 + j0) * 4;
  const float* u2b = U2 + ((size_t)zz * 64 + j1) * 4;
  const float* u3a = U3 + ((size_t)zz * 64 + j0) * 4;
  const float* u3b = U3 + ((size_t)zz * 64 + j1) * 4;
  float acc0 = 0.f, acc1 = 0.f;
#pragma unroll
  for (int l = 0; l < 4; ++l) {
    acc0 += (u2a[l] + i10 * u3a[l]) * p0[l];
    acc1 += (u2b[l] + i11 * u3b[l]) * p1[l];
  }
  const float* WscZ = Wsc + (size_t)zz * 4096;
  const float* inv1row = Amix + base;  // m = 0 slice
  const float* frow = feats + base;
  for (int k = 0; k < 64; ++k) {
    const float fk = frow[k];
    const float ik = inv1row[k];
    acc0 += fk * WscZ[k * 64 + j0] + ik * Winv1[k * 64 + j0];
    acc1 += fk * WscZ[k * 64 + j1] + ik * Winv1[k * 64 + j1];
  }
  feats[base + j0] = acc0;
  feats[base + j1] = acc1;
  sF[w][j0] = acc0;
  sF[w][j1] = acc1;
  __builtin_amdgcn_wave_barrier();

  float es;
  if (is_last) {
    float part = 0.f;
    if (lane < 16) {
      float h = 0.f;
      for (int k = 0; k < 64; ++k) h += sF[w][k] * Wr1[k * 16 + lane];
      part = silu(h) * Wr2[lane];
    }
    es = part;
  } else {
    es = acc0 * Wread0[j0] + acc1 * Wread0[j1];
  }
#pragma unroll
  for (int off = 16; off > 0; off >>= 1) es += __shfl_down(es, off, 32);
  if (lane == 0) atomicAdd(&energy[batch[n]], es);
}

// ---------------------------------------------------------------------------
extern "C" void kernel_launch(void* const* d_in, const int* in_sizes, int n_in,
                              void* d_out, int out_size, void* d_ws, size_t ws_size,
                              hipStream_t stream) {
  (void)n_in; (void)ws_size;
  const float* vectors = (const float*)d_in[0];
  const float* attrs   = (const float*)d_in[1];
  const int* eidx      = (const int*)d_in[2];
  const int* batch     = (const int*)d_in[3];
  const float* W_embed = (const float*)d_in[4];
  const float* ae_w    = (const float*)d_in[5];
  const float* R0      = (const float*)d_in[6];
  const float* R1      = (const float*)d_in[7];
  const float* R2      = (const float*)d_in[8];
  const float* R3      = (const float*)d_in[9];
  const float* Wmix    = (const float*)d_in[10];
  const float* Wsc     = (const float*)d_in[11];
  const float* Winv1   = (const float*)d_in[12];
  const float* U2      = (const float*)d_in[13];
  const float* U3      = (const float*)d_in[14];
  const float* Wread0  = (const float*)d_in[15];
  const float* Wr1     = (const float*)d_in[16];
  const float* Wr2     = (const float*)d_in[17];
  float* energy = (float*)d_out;

  const int E = in_sizes[0] / 3;
  const int N = in_sizes[1] / 10;
  const int* sender = eidx;
  const int* recv = eidx + E;

  char* ws = (char*)d_ws;
  size_t off = 0;
  auto take = [&](size_t bytes) -> void* {
    void* p = ws + off;
    off += (bytes + 255) & ~(size_t)255;
    return p;
  };
  float* Yg   = (float*)take((size_t)E * 16 * sizeof(float));
  bf16* efb   = (bf16*)take((size_t)E * 8 * sizeof(bf16));
  float* feats = (float*)take((size_t)N * 64 * sizeof(float));
  int* zbuf   = (int*)take((size_t)N * sizeof(int));
  float* Apre = (float*)take((size_t)16 * N * 64 * sizeof(float));
  float* Amix = (float*)take((size_t)16 * N * 64 * sizeof(float));

  // energy = 0
  mace_zero4_kernel<<<1, 64, 0, stream>>>((float4*)energy, (size_t)out_size / 4);
  // z, feats0, atom baseline energies
  mace_setup_kernel<<<(N * 64 + 255) / 256, 256, 0, stream>>>(
      attrs, W_embed, ae_w, batch, feats, zbuf, energy, N);
  // geometry
  mace_geom_kernel<<<(E + 255) / 256, 256, 0, stream>>>(vectors, Yg, efb, E);

  const size_t nA4 = (size_t)16 * N * 64 / 4;
  const int tiles = E / 16;
  for (int i = 0; i < 2; ++i) {
    mace_zero4_kernel<<<2048, 256, 0, stream>>>((float4*)Apre, nA4);
    mace_edge_kernel<<<(tiles + 3) / 4, 128, 0, stream>>>(
        R0 + (size_t)i * 8 * 64, R1 + (size_t)i * 64 * 64, R2 + (size_t)i * 64 * 64,
        R3 + (size_t)i * 64 * 256, efb, Yg, feats, sender, recv, Apre, E, N);
    mace_mix_kernel<<<dim3(((N + 15) / 16 + 7) / 8, 16), 256, 0, stream>>>(
        Apre, Wmix + (size_t)i * 4 * 64 * 64, Amix, N);
    mace_node_kernel<<<(N + 7) / 8, 256, 0, stream>>>(
        Amix, feats, zbuf, batch, U2 + (size_t)i * 10 * 64 * 4, U3 + (size_t)i * 10 * 64 * 4,
        Wsc + (size_t)i * 10 * 64 * 64, Winv1 + (size_t)i * 64 * 64,
        Wread0, Wr1, Wr2, energy, N, (i == 1) ? 1 : 0);
  }
}
